// MSA_52029233823832
// MI455X (gfx1250) — matrix operand order repaired and checked by reading, not compiled, not run on hardware
//
#include <hip/hip_runtime.h>

// ---------------------------------------------------------------------------
// Types
// ---------------------------------------------------------------------------
typedef __bf16 bf16_t;
typedef __attribute__((ext_vector_type(4)))  bf16_t v4bf;
typedef __attribute__((ext_vector_type(8)))  bf16_t v8bf;
typedef __attribute__((ext_vector_type(16))) bf16_t v16bf;
typedef __attribute__((ext_vector_type(8)))  float  v8f;
typedef __attribute__((ext_vector_type(4)))  int    v4i;
typedef __attribute__((ext_vector_type(8)))  int    v8i;
typedef __attribute__((ext_vector_type(4)))  unsigned int v4u;

#define AS1 __attribute__((address_space(1)))
#define AS3 __attribute__((address_space(3)))

#if __has_builtin(__builtin_amdgcn_global_load_async_to_lds_b128)
#define HAVE_ASYNC_LDS 1
#else
#define HAVE_ASYNC_LDS 0
#endif

#if __has_builtin(__builtin_amdgcn_tensor_load_to_lds)
#define HAVE_TDM 1
#else
#define HAVE_TDM 0
#endif

__device__ __forceinline__ v8f v8f_zero() {
    v8f v;
#pragma unroll
    for (int i = 0; i < 8; ++i) v[i] = 0.0f;
    return v;
}

// Build a 16-element bf16 fragment from two 16-byte chunks.
__device__ __forceinline__ v16bf make_frag(const bf16_t* p0, const bf16_t* p1) {
    v8bf lo = *(const v8bf*)p0;
    v8bf hi = *(const v8bf*)p1;
    return __builtin_shufflevector(lo, hi, 0, 1, 2, 3, 4, 5, 6, 7,
                                           8, 9, 10, 11, 12, 13, 14, 15);
}

__device__ __forceinline__ v8f wmma_bf16(v16bf a, v16bf b, v8f c) {
    // (neg_a, A, neg_b, B, c_mod, C, reuse_a, reuse_b)
    return __builtin_amdgcn_wmma_f32_16x16x32_bf16(false, a, false, b,
                                                   (short)0, c, false, false);
}

// 16-byte global -> LDS copy (async on CDNA5; sync fallback otherwise).
__device__ __forceinline__ void cp16_g2l(const bf16_t* g, bf16_t* l) {
#if HAVE_ASYNC_LDS
    __builtin_amdgcn_global_load_async_to_lds_b128(
        (AS1 v4i*)g, (AS3 v4i*)l, 0, 0);
#else
    *(v8bf*)l = *(const v8bf*)g;
#endif
}

__device__ __forceinline__ void wait_async_done() {
#if HAVE_ASYNC_LDS
#if __has_builtin(__builtin_amdgcn_s_wait_asynccnt)
    __builtin_amdgcn_s_wait_asynccnt(0);
#else
    asm volatile("s_wait_asynccnt 0x0" ::: "memory");
#endif
#endif
}

__device__ __forceinline__ void wait_tensor_done() {
#if __has_builtin(__builtin_amdgcn_s_wait_tensorcnt)
    __builtin_amdgcn_s_wait_tensorcnt(0);
#else
    asm volatile("s_wait_tensorcnt 0x0" ::: "memory");
#endif
}

#if HAVE_TDM
// Issue a TDM 2-D tile load: 128 rows x 32 bf16 (64B) per row into LDS with
// a 16B pad after every row (LDS pitch 80B). D# layout per CDNA5 ISA 8.3-8.6.
__device__ __forceinline__ void tdm_load_tile_128x32(
    const bf16_t* gtile,   // global address of tile start
    unsigned lds_off,      // byte offset of destination in LDS
    unsigned row_len,      // tensor dim0 (elements)
    unsigned row_stride)   // tensor dim0 stride (elements)
{
    const unsigned long long ga = (unsigned long long)(size_t)gtile;
    v4u g0;
    g0[0] = 1u;                                   // count=1, user descriptor
    g0[1] = lds_off;                              // lds_addr (bytes)
    g0[2] = (unsigned)(ga & 0xffffffffu);         // global_addr[31:0]
    g0[3] = (unsigned)((ga >> 32) & 0x1ffffffu)   // global_addr[56:32]
          | (2u << 30);                           // type = 2 ("image")
    v8i g1;
    g1[0] = (1 << 16)                             // data_size = 2 bytes
          | (1 << 20)                             // pad_enable
          | (3 << 22)                             // pad_interval: 16 DWORDs (64B)
          | (3 << 25);                            // pad_amount: 4 DWORDs (16B)
    g1[1] = (int)((row_len & 0xffffu) << 16);     // tensor_dim0[15:0]
    g1[2] = (int)((row_len >> 16) | (0x0u << 16)) // tensor_dim0[31:16] | dim1 lo
          | (int)((0x00100000u & 0xffffu) << 16); // tensor_dim1[15:0] (=0 low bits)
    g1[3] = (int)((0x00100000u >> 16))            // tensor_dim1[31:16] (1M rows)
          | (32 << 16);                           // tile_dim0 = 32
    g1[4] = 128;                                  // tile_dim1 = 128, tile_dim2 = 0
    g1[5] = (int)row_stride;                      // tensor_dim0_stride[31:0]
    g1[6] = 0;                                    // stride hi / dim1_stride lo
    g1[7] = 0;                                    // dim1_stride hi
    v4i gz4 = {0, 0, 0, 0};                       // groups 2/3 unused (2-D)
    v8i gz8 = {0, 0, 0, 0, 0, 0, 0, 0};           // extra group (clang-23 form)
    __builtin_amdgcn_tensor_load_to_lds(g0, g1, gz4, gz4, gz8, 0);
}
#endif

// ---------------------------------------------------------------------------
// f32 -> bf16 conversion (vec4)
// ---------------------------------------------------------------------------
__global__ void cvt_f32_bf16(const float* __restrict__ in,
                             bf16_t* __restrict__ out, int n) {
    int i = (blockIdx.x * blockDim.x + threadIdx.x) * 4;
    if (i + 3 < n) {
        float4 v = *(const float4*)(in + i);
        v4bf o;
        o[0] = (bf16_t)v.x; o[1] = (bf16_t)v.y;
        o[2] = (bf16_t)v.z; o[3] = (bf16_t)v.w;
        *(v4bf*)(out + i) = o;
    }
}

// ---------------------------------------------------------------------------
// Core 128x128 GEMM tile: C = A[M,K] * B[N,K]^T  (bf16 in, f32 acc)
// 256 threads = 8 waves; wave w computes rows [w*16, w*16+16) x 128 cols.
// Double-buffered LDS tiles (padded to 40 bf16/row). Tile i+1 is fetched by
// the Tensor Data Mover (wave 0 issues, TENSORcnt tracks) while all waves run
// WMMAs on tile i; falls back to per-lane async-to-LDS copies without TDM.
// ---------------------------------------------------------------------------
__device__ __forceinline__ void gemm_128x128(
    const bf16_t* __restrict__ A, const bf16_t* __restrict__ B, int K,
    int mBase, int nBase, v8f acc[8],
    bf16_t (*Alds)[128][40], bf16_t (*Blds)[128][40])
{
    const int tid  = threadIdx.x;
    const int wave = tid >> 5;
    const int lane = tid & 31;
    const int fr   = lane & 15;   // fragment row (A) / col (B)
    const int hi   = lane >> 4;   // lane half selects contract sub-range
    const int ldRow   = tid >> 1;       // 0..127
    const int ldChunk = (tid & 1) * 2;  // chunks {0,1} or {2,3} of 8 bf16

#pragma unroll
    for (int i = 0; i < 8; ++i) acc[i] = v8f_zero();

    const int nk = K >> 5;  // K/32 k-steps

#if HAVE_TDM
    const unsigned ldsA0 = (unsigned)(size_t)(AS3 void*)&Alds[0][0][0];
    const unsigned ldsA1 = (unsigned)(size_t)(AS3 void*)&Alds[1][0][0];
    const unsigned ldsB0 = (unsigned)(size_t)(AS3 void*)&Blds[0][0][0];
    const unsigned ldsB1 = (unsigned)(size_t)(AS3 void*)&Blds[1][0][0];
    if (wave == 0) {
        tdm_load_tile_128x32(A + (size_t)mBase * K, ldsA0, (unsigned)K, (unsigned)K);
        tdm_load_tile_128x32(B + (size_t)nBase * K, ldsB0, (unsigned)K, (unsigned)K);
    }
#else
    {
        const bf16_t* ga = A + (size_t)(mBase + ldRow) * K + ldChunk * 8;
        cp16_g2l(ga,     &Alds[0][ldRow][ldChunk * 8]);
        cp16_g2l(ga + 8, &Alds[0][ldRow][ldChunk * 8 + 8]);
        const bf16_t* gb = B + (size_t)(nBase + ldRow) * K + ldChunk * 8;
        cp16_g2l(gb,     &Blds[0][ldRow][ldChunk * 8]);
        cp16_g2l(gb + 8, &Blds[0][ldRow][ldChunk * 8 + 8]);
    }
#endif

    for (int i = 0; i < nk; ++i) {
        const int cur = i & 1;
#if HAVE_TDM
        if (wave == 0) wait_tensor_done();  // wave 0's DMA for tile i landed
#else
        wait_async_done();                  // this wave's copies for tile i landed
#endif
        __syncthreads();  // tile i visible to all; all done reading buf cur^1

        if (i + 1 < nk) {  // overlap tile i+1 fetch with tile i compute
            const int k0 = (i + 1) * 32;
#if HAVE_TDM
            if (wave == 0) {
                tdm_load_tile_128x32(A + (size_t)mBase * K + k0,
                                     cur ? ldsA0 : ldsA1, (unsigned)K, (unsigned)K);
                tdm_load_tile_128x32(B + (size_t)nBase * K + k0,
                                     cur ? ldsB0 : ldsB1, (unsigned)K, (unsigned)K);
            }
#else
            const int nxt = cur ^ 1;
            const bf16_t* ga = A + (size_t)(mBase + ldRow) * K + k0 + ldChunk * 8;
            cp16_g2l(ga,     &Alds[nxt][ldRow][ldChunk * 8]);
            cp16_g2l(ga + 8, &Alds[nxt][ldRow][ldChunk * 8 + 8]);
            const bf16_t* gb = B + (size_t)(nBase + ldRow) * K + k0 + ldChunk * 8;
            cp16_g2l(gb,     &Blds[nxt][ldRow][ldChunk * 8]);
            cp16_g2l(gb + 8, &Blds[nxt][ldRow][ldChunk * 8 + 8]);
#endif
        }

        // A fragment: row = wave*16 + fr, contract {hi*8..+7, hi*8+16..+23}
        v16bf afrag = make_frag(&Alds[cur][wave * 16 + fr][hi * 8],
                                &Alds[cur][wave * 16 + fr][hi * 8 + 16]);
        // Batch all B fragments, then issue the 8 WMMAs back-to-back.
        v16bf bfr[8];
#pragma unroll
        for (int t = 0; t < 8; ++t)
            bfr[t] = make_frag(&Blds[cur][t * 16 + fr][hi * 8],
                               &Blds[cur][t * 16 + fr][hi * 8 + 16]);
#pragma unroll
        for (int t = 0; t < 8; ++t)
            acc[t] = wmma_bf16(afrag, bfr[t], acc[t]);
    }
}

// ---------------------------------------------------------------------------
// Stage 1: QKV = X @ Wqkv^T ; scatter into q (scaled), k, v^T (bf16)
// X: [8192,1024], Wqkv: [3072,1024]. grid (24, 64) x 256.
// ---------------------------------------------------------------------------
__global__ __launch_bounds__(256) void qkv_gemm_kernel(
    const bf16_t* __restrict__ xb, const bf16_t* __restrict__ wqkvb,
    bf16_t* __restrict__ qb, bf16_t* __restrict__ kb, bf16_t* __restrict__ vtb)
{
    __shared__ __align__(16) bf16_t Alds[2][128][40];
    __shared__ __align__(16) bf16_t Blds[2][128][40];
    const int mBase = blockIdx.y * 128;
    const int nBase = blockIdx.x * 128;
    v8f acc[8];
    gemm_128x128(xb, wqkvb, 1024, mBase, nBase, acc, Alds, Blds);

    const int tid = threadIdx.x, wave = tid >> 5, lane = tid & 31;
    const int fr = lane & 15, hi = lane >> 4;
#pragma unroll
    for (int t = 0; t < 8; ++t) {
#pragma unroll
        for (int r = 0; r < 8; ++r) {
            const int m = mBase + wave * 16 + r + hi * 8;   // token index b*2048+n
            const int f = nBase + t * 16 + fr;              // 0..3071
            const float val = acc[t][r];
            const int b = m >> 11, n = m & 2047;
            const int sec = f >> 10, e = f & 1023;
            const int h = e >> 6, d = e & 63;
            const size_t bh = (size_t)(b * 16 + h);
            if (sec == 0)
                qb[(bh * 2048 + n) * 64 + d] = (bf16_t)(val * 0.125f); // /sqrt(64)
            else if (sec == 1)
                kb[(bh * 2048 + n) * 64 + d] = (bf16_t)val;
            else
                vtb[(bh * 64 + d) * 2048 + n] = (bf16_t)val;           // transposed
        }
    }
}

// ---------------------------------------------------------------------------
// Stage 2: flash attention. grid 1024 = (b*16+h)*16 row-tiles; 8 waves/block,
// each wave owns 16 query rows; loops over 32-key tiles of N=2048.
// ---------------------------------------------------------------------------
__global__ __launch_bounds__(256) void attn_kernel(
    const bf16_t* __restrict__ qb, const bf16_t* __restrict__ kb,
    const bf16_t* __restrict__ vtb, float* __restrict__ sv_out,
    bf16_t* __restrict__ svb)
{
    __shared__ __align__(16) bf16_t Plds[8][16][40];   // per-wave P tile 16x32 (+pad)
    const int tid = threadIdx.x, wave = tid >> 5, lane = tid & 31;
    const int fr = lane & 15, hi = lane >> 4;
    const int blk   = blockIdx.x;
    const int qTile = (blk & 15) * 128;
    const int bh    = blk >> 4;                         // b*16 + h
    const bf16_t* Q  = qb  + (size_t)bh * 2048 * 64;
    const bf16_t* Kp = kb  + (size_t)bh * 2048 * 64;
    const bf16_t* Vt = vtb + (size_t)bh * 64 * 2048;
    const int qRow = qTile + wave * 16;

    // Q fragments for the two 32-wide d-chunks (hd = 64)
    v16bf qf[2];
#pragma unroll
    for (int c = 0; c < 2; ++c) {
        const bf16_t* p = Q + (size_t)(qRow + fr) * 64 + c * 32 + hi * 8;
        qf[c] = make_frag(p, p + 16);
    }

    float mrow[8], lrow[8];
    v8f acc[4];
#pragma unroll
    for (int r = 0; r < 8; ++r) { mrow[r] = -1e30f; lrow[r] = 0.0f; }
#pragma unroll
    for (int t = 0; t < 4; ++t) acc[t] = v8f_zero();

    for (int kt = 0; kt < 2048; kt += 32) {
        // ---- S = Q K^T for 32 keys: batch 4 K fragments, then 4 WMMAs ----
        v16bf kf[4];
#pragma unroll
        for (int g = 0; g < 2; ++g)
#pragma unroll
            for (int c = 0; c < 2; ++c) {
                const bf16_t* p =
                    Kp + (size_t)(kt + g * 16 + fr) * 64 + c * 32 + hi * 8;
                kf[g * 2 + c] = make_frag(p, p + 16);
            }
        v8f s0 = v8f_zero(), s1 = v8f_zero();
        s0 = wmma_bf16(qf[0], kf[0], s0);
        s0 = wmma_bf16(qf[1], kf[1], s0);
        s1 = wmma_bf16(qf[0], kf[2], s1);
        s1 = wmma_bf16(qf[1], kf[3], s1);

        // ---- online softmax; rows r (lanes 0-15) and r+8 (lanes 16-31) ----
#pragma unroll
        for (int r = 0; r < 8; ++r) {
            float mx = fmaxf(s0[r], s1[r]);
#pragma unroll
            for (int off = 8; off; off >>= 1)
                mx = fmaxf(mx, __shfl_xor(mx, off, 16));
            const float newm = fmaxf(mrow[r], mx);
            const float corr = __expf(mrow[r] - newm);
            mrow[r] = newm;
            const float p0v = __expf(s0[r] - newm);
            const float p1v = __expf(s1[r] - newm);
            float rs = p0v + p1v;
#pragma unroll
            for (int off = 8; off; off >>= 1)
                rs += __shfl_xor(rs, off, 16);
            lrow[r] = lrow[r] * corr + rs;
#pragma unroll
            for (int t = 0; t < 4; ++t) acc[t][r] *= corr;
            // P (bf16) into per-wave LDS tile, C-layout -> [row][key] matrix
            Plds[wave][r + hi * 8][fr]      = (bf16_t)p0v;
            Plds[wave][r + hi * 8][fr + 16] = (bf16_t)p1v;
        }
        // same-wave LDS store -> load fence
        asm volatile("s_wait_dscnt 0x0" ::: "memory");
        // ---- reload P as A-fragment (16x32 over keys) ----
        const bf16_t* pp = &Plds[wave][fr][hi * 8];
        v16bf pf = make_frag(pp, pp + 16);
        // ---- O += P @ V : batch 4 V^T fragments, then 4 WMMAs ----
        v16bf vf[4];
#pragma unroll
        for (int t = 0; t < 4; ++t) {
            const bf16_t* vp = Vt + (size_t)(t * 16 + fr) * 2048 + kt + hi * 8;
            vf[t] = make_frag(vp, vp + 16);
        }
#pragma unroll
        for (int t = 0; t < 4; ++t)
            acc[t] = wmma_bf16(pf, vf[t], acc[t]);
    }

    // ---- epilogue: sv = O / l, write f32 output + bf16 copy for proj ----
    const int b = bh >> 4, h = bh & 15;
#pragma unroll
    for (int t = 0; t < 4; ++t) {
#pragma unroll
        for (int r = 0; r < 8; ++r) {
            const int n = qRow + r + hi * 8;
            const int e = h * 64 + t * 16 + fr;
            const float val = acc[t][r] / lrow[r];
            const size_t idx = ((size_t)b * 2048 + n) * 1024 + e;
            sv_out[idx] = val;
            svb[idx]    = (bf16_t)val;
        }
    }
}

// ---------------------------------------------------------------------------
// Stage 3: out = sv @ Wproj^T ; grid (8, 64) x 256
// ---------------------------------------------------------------------------
__global__ __launch_bounds__(256) void proj_gemm_kernel(
    const bf16_t* __restrict__ svb, const bf16_t* __restrict__ wprojb,
    float* __restrict__ out)
{
    __shared__ __align__(16) bf16_t Alds[2][128][40];
    __shared__ __align__(16) bf16_t Blds[2][128][40];
    const int mBase = blockIdx.y * 128;
    const int nBase = blockIdx.x * 128;
    v8f acc[8];
    gemm_128x128(svb, wprojb, 1024, mBase, nBase, acc, Alds, Blds);

    const int tid = threadIdx.x, wave = tid >> 5, lane = tid & 31;
    const int fr = lane & 15, hi = lane >> 4;
#pragma unroll
    for (int t = 0; t < 8; ++t) {
#pragma unroll
        for (int r = 0; r < 8; ++r) {
            const int m = mBase + wave * 16 + r + hi * 8;
            const int n = nBase + t * 16 + fr;
            out[(size_t)m * 1024 + n] = acc[t][r];
        }
    }
}

// ---------------------------------------------------------------------------
// Launch
// ---------------------------------------------------------------------------
extern "C" void kernel_launch(void* const* d_in, const int* in_sizes, int n_in,
                              void* d_out, int out_size, void* d_ws, size_t ws_size,
                              hipStream_t stream)
{
    const float* x     = (const float*)d_in[0];   // [4,2048,1024]
    const float* Wqkv  = (const float*)d_in[1];   // [3072,1024]
    const float* Wproj = (const float*)d_in[2];   // [1024,1024]
    float* out    = (float*)d_out;                      // [8192*1024]
    float* sv_out = out + (size_t)8192 * 1024;          // second output

    const size_t MN  = (size_t)8192 * 1024;   // tokens x E
    const size_t QKN = (size_t)3072 * 1024;
    const size_t PN  = (size_t)1024 * 1024;
    const size_t HBN = (size_t)64 * 2048 * 64; // B*H x N x hd

    bf16_t* xb     = (bf16_t*)d_ws;
    bf16_t* wqkvb  = xb + MN;
    bf16_t* wprojb = wqkvb + QKN;
    bf16_t* qb     = wprojb + PN;
    bf16_t* kb     = qb + HBN;
    bf16_t* vtb    = kb + HBN;
    bf16_t* svb    = vtb + HBN;

    // Stage 0: f32 -> bf16 converts
    {
        int n = (int)MN;
        cvt_f32_bf16<<<(n / 4 + 255) / 256, 256, 0, stream>>>(x, xb, n);
        n = (int)QKN;
        cvt_f32_bf16<<<(n / 4 + 255) / 256, 256, 0, stream>>>(Wqkv, wqkvb, n);
        n = (int)PN;
        cvt_f32_bf16<<<(n / 4 + 255) / 256, 256, 0, stream>>>(Wproj, wprojb, n);
    }
    // Stage 1: QKV GEMM (8192x3072x1024)
    qkv_gemm_kernel<<<dim3(24, 64), 256, 0, stream>>>(xb, wqkvb, qb, kb, vtb);
    // Stage 2: flash attention (B*H=64 heads, 16 row-tiles each)
    attn_kernel<<<1024, 256, 0, stream>>>(qb, kb, vtb, sv_out, svb);
    // Stage 3: projection GEMM (8192x1024x1024)
    proj_gemm_kernel<<<dim3(8, 64), 256, 0, stream>>>(svb, wprojb, out);
}
